// VectorizedQuantumGeometricAttention_50122268344536
// MI455X (gfx1250) — compile-verified, matrix operand order checked by reading, not run on hardware
//
#include <hip/hip_runtime.h>
#include <hip/hip_bf16.h>

// VectorizedQuantumGeometricAttention for MI455X (gfx1250, wave32, WMMA).
//
// embeddings: [B=8, S=2048, D=256] f32.
// attn[b,s,t] = softmax_t( -dist(b,s,t) ), dist = sqrt(clip(|e_s|^2+|e_t|^2-2<e_s,e_t>, 1e-12)),
// diagonal dist forced to 0. Since dist>=0 and diag==0, row max of -dist is 0,
// so softmax = exp(-dist)/sum(exp(-dist)) -- single pass, no max sweep.
//
// Roofline: 17.2 GFLOP GEMM, 128 MiB output -> ~6us HBM floor @23.3TB/s.
// bf16 WMMA (16 KFLOP/instr) keeps the GEMM at/below the memory floor;
// fp32-native 16x16x4 WMMA (2 KFLOP/instr) would be ~8x too slow.
// bf16 dot error ~3e-3 absolute on dist ~22.6 -> ~0.3% softmax error.

#define BB 8
#define SS 2048
#define DD 256
#define STRIDE 2052   // 32*2052*4 = 256.5 KB LDS strip, padded against bank conflicts

typedef __attribute__((ext_vector_type(16))) __bf16 v16bf;
typedef __attribute__((ext_vector_type(8)))  float  v8f;

union BF16Frag { uint4 q[2]; v16bf v; };

__device__ __forceinline__ unsigned short f2bf_rne(float f) {
    union { float f; unsigned u; } c; c.f = f;
    unsigned u = c.u + 0x7FFFu + ((c.u >> 16) & 1u);
    return (unsigned short)(u >> 16);
}

// ---------------- Kernel 1: bf16 convert + exact fp32 row norms ----------------
// grid = B*S blocks of 32 (one wave per row of 256 floats)
__global__ void qga_prep(const float* __restrict__ E,
                         unsigned short* __restrict__ ehi,
                         float* __restrict__ norms) {
    const int row  = blockIdx.x;
    const int lane = threadIdx.x;            // 0..31
    const float* src = E + (size_t)row * DD;
    unsigned short* dst = ehi + (size_t)row * DD;

    float acc = 0.0f;
#pragma unroll
    for (int i = 0; i < 2; ++i) {
        const int idx = (i * 32 + lane) * 4; // float4 granularity, coalesced
        float4 v = *(const float4*)(src + idx);
        acc += v.x * v.x + v.y * v.y + v.z * v.z + v.w * v.w;
        ushort4 h;
        h.x = f2bf_rne(v.x); h.y = f2bf_rne(v.y);
        h.z = f2bf_rne(v.z); h.w = f2bf_rne(v.w);
        *(ushort4*)(dst + idx) = h;
    }
#pragma unroll
    for (int off = 16; off >= 1; off >>= 1) acc += __shfl_xor(acc, off, 32);
    if (lane == 0) norms[row] = acc;
}

// ---------------- Kernel 2: fused Gram + dist + exp + softmax ----------------
// grid = (S/32, B), block = 256 (8 waves). WG owns a 32-row strip.
// Each wave: A-panel resident in VGPRs (32 rows x K=256 -> 16 bf16 fragments),
// sweeps its 256-column slice; exp(-dist) staged in LDS; fused normalize+store.
__launch_bounds__(256, 1)
__global__ void qga_attn(const unsigned short* __restrict__ ehi,
                         const float* __restrict__ norms,
                         float* __restrict__ out) {
    __shared__ float strip[32 * STRIDE];
    __shared__ float rown[32];

    const int b    = blockIdx.y;
    const int r0   = blockIdx.x * 32;        // strip base row
    const int tid  = threadIdx.x;
    const int wave = tid >> 5;               // 0..7
    const int lane = tid & 31;
    const int half = lane >> 4;              // 0 or 1
    const int l15  = lane & 15;

    const unsigned short* Eb = ehi + (size_t)b * SS * DD;
    const float* Nb = norms + (size_t)b * SS;

    if (tid < 32) rown[tid] = Nb[r0 + tid];
    __syncthreads();

    // Resident A panel: 2 M-tiles x 8 K-steps. Per-lane layout for 16-bit A 16x32:
    // lanes 0-15: row M=l, K = {0..7,16..23}; lanes 16-31: row M=l-16, K = {8..15,24..31}.
    BF16Frag afrag[2][8];
#pragma unroll
    for (int mt = 0; mt < 2; ++mt) {
        const unsigned short* abase = Eb + (size_t)(r0 + mt * 16 + l15) * DD;
#pragma unroll
        for (int ks = 0; ks < 8; ++ks) {
            const int k0 = ks * 32 + half * 8;
            afrag[mt][ks].q[0] = *(const uint4*)(abase + k0);
            afrag[mt][ks].q[1] = *(const uint4*)(abase + k0 + 16);
        }
    }

    const int cb_wave = wave * 256;          // this wave's column slice
    for (int nt = 0; nt < 16; ++nt) {
        const int cb = cb_wave + nt * 16;
        // For the Gram matrix, the B fragment (col-per-lane, K packed like A)
        // is loaded identically to an A fragment of the column rows.
        const unsigned short* bbase = Eb + (size_t)(cb + l15) * DD;

        v8f c0 = {}; v8f c1 = {};
#pragma unroll
        for (int ks = 0; ks < 8; ++ks) {
            BF16Frag bf;
            const int k0 = ks * 32 + half * 8;
            bf.q[0] = *(const uint4*)(bbase + k0);
            bf.q[1] = *(const uint4*)(bbase + k0 + 16);
            c0 = __builtin_amdgcn_wmma_f32_16x16x32_bf16(
                     false, afrag[0][ks].v, false, bf.v, (short)0, c0, false, false);
            c1 = __builtin_amdgcn_wmma_f32_16x16x32_bf16(
                     false, afrag[1][ks].v, false, bf.v, (short)0, c1, false, false);
        }

        // Epilogue: sq_dist -> sqrt -> diag zero -> exp, into LDS strip.
        // C/D layout: VGPR i, lanes 0-15 -> M=i, lanes 16-31 -> M=i+8; N=lane&15.
        const int col = cb + l15;
        const float nc = Nb[col];
#pragma unroll
        for (int i = 0; i < 8; ++i) {
            const int rl0 = i + half * 8;            // local row for m-tile 0
            float d0 = rown[rl0] + nc - 2.0f * c0[i];
            d0 = __builtin_sqrtf(fmaxf(d0, 1e-12f));
            if (r0 + rl0 == col) d0 = 0.0f;
            strip[rl0 * STRIDE + col] = __expf(-d0);

            const int rl1 = rl0 + 16;                // local row for m-tile 1
            float d1 = rown[rl1] + nc - 2.0f * c1[i];
            d1 = __builtin_sqrtf(fmaxf(d1, 1e-12f));
            if (r0 + rl1 == col) d1 = 0.0f;
            strip[rl1 * STRIDE + col] = __expf(-d1);
        }
    }
    __syncthreads();

    // Row softmax: wave w normalizes rows 4w..4w+3 (sum over 2048 cols, then
    // fused scale + coalesced float4 stores to HBM).
#pragma unroll
    for (int rr4 = 0; rr4 < 4; ++rr4) {
        const int rr = wave * 4 + rr4;
        const float4* rowp = (const float4*)&strip[rr * STRIDE]; // 16B aligned (2052*4 % 16 == 0)
        float s = 0.0f;
        for (int j = lane; j < SS / 4; j += 32) {
            float4 v = rowp[j];
            s += (v.x + v.y) + (v.z + v.w);
        }
#pragma unroll
        for (int off = 16; off >= 1; off >>= 1) s += __shfl_xor(s, off, 32);
        const float inv = 1.0f / s;

        float4* orow = (float4*)(out + ((size_t)b * SS + (r0 + rr)) * SS);
        for (int j = lane; j < SS / 4; j += 32) {
            float4 v = rowp[j];
            v.x *= inv; v.y *= inv; v.z *= inv; v.w *= inv;
            orow[j] = v;
        }
    }
}

extern "C" void kernel_launch(void* const* d_in, const int* in_sizes, int n_in,
                              void* d_out, int out_size, void* d_ws, size_t ws_size,
                              hipStream_t stream) {
    const float* E = (const float*)d_in[0];
    float* out = (float*)d_out;

    // Workspace layout: [bf16 embeddings: B*S*D*2 = 8 MiB][row norms: B*S*4 = 64 KB]
    unsigned short* ehi = (unsigned short*)d_ws;
    float* norms = (float*)((char*)d_ws + (size_t)BB * SS * DD * 2);

    qga_prep<<<BB * SS, 32, 0, stream>>>(E, ehi, norms);

    dim3 grid(SS / 32, BB);
    qga_attn<<<grid, 256, 0, stream>>>(ehi, norms, out);
}
